// nodal_MLP_9603546873894
// MI455X (gfx1250) — matrix operand order repaired and checked by reading, not compiled, not run on hardware
//
#include <hip/hip_runtime.h>

typedef __attribute__((ext_vector_type(16))) __bf16 v16bf;
typedef __attribute__((ext_vector_type(8)))  __bf16 v8bf;
typedef __attribute__((ext_vector_type(8)))  float  v8f;
typedef __attribute__((ext_vector_type(4)))  unsigned int v4u;
typedef __attribute__((ext_vector_type(8)))  int v8i;
typedef __attribute__((ext_vector_type(4)))  int v4i;

#define BLOCK  256
#define WAVES  8
#define NT     7      // 112 = 7 * 16 output tiles
#define HREAL  100
#define HPAD   112
#define KPAD2  128    // K padding for layers 2/3 (4 * 32)
#define AROWS  64     // input features (edges)

// d_ws layout (bytes)
#define WF1_OFF   0          // 7*2*512 bf16 = 14336 B
#define WF2_OFF   14336      // 7*4*512 bf16 = 28672 B
#define WF3_OFF   43008      // 7*4*512 bf16 = 28672 B
#define FTAB_OFF  71680      // f32: gate[64], b123[336], w4p[112], b4[1]
#define ITAB_OFF  73984      // int: gidx[64]

struct A16 { __bf16 e[16]; };

// round-to-nearest-even (used only in the one-shot prep kernel)
__device__ __forceinline__ __bf16 f2bf_rne(float f) {
  unsigned u = __builtin_bit_cast(unsigned, f);
  unsigned r = (u + 0x7FFFu + ((u >> 16) & 1u)) >> 16;
  unsigned short s = (unsigned short)r;
  return __builtin_bit_cast(__bf16, s);
}
// truncation (hot path: folds into d16_hi moves/stores, ~free)
__device__ __forceinline__ __bf16 f2bf_t(float f) {
  unsigned u = __builtin_bit_cast(unsigned, f);
  unsigned short s = (unsigned short)(u >> 16);
  return __builtin_bit_cast(__bf16, s);
}
__device__ __forceinline__ float bf2f(__bf16 b) {
  unsigned short s = __builtin_bit_cast(unsigned short, b);
  unsigned u = ((unsigned)s) << 16;
  return __builtin_bit_cast(float, u);
}

// ---------------------------------------------------------------------------
// Prep kernel (1 block, runs once per launch): convert f32 weights into bf16
// B-fragment-linear blobs + small f32/int tables in d_ws.
// Fragment (nt, ks): K = ks*32..+31, N = nt*16..+15; lane L owns col
// nt*16+(L&15), K values (L>>4)*8 + i (i<8), 16 + (L>>4)*8 + (i-8) (i>=8) -> 2x ds_load_b128 per use.
// ---------------------------------------------------------------------------
__device__ __forceinline__ __bf16 frag_elem(const float* __restrict__ W,
                                            int Kreal, int ksteps, int e) {
  int frag = e >> 9;
  int rem  = e & 511;
  int ln   = rem >> 4;
  int i    = rem & 15;
  int nt   = frag / ksteps;
  int ks   = frag - nt * ksteps;
  int col  = nt * 16 + (ln & 15);
  int krow = ks * 32 + ((i < 8) ? ((ln >> 4) * 8 + i) : (16 + (ln >> 4) * 8 + (i - 8)));
  float v = 0.0f;
  if (krow < Kreal && col < HREAL) v = W[krow * HREAL + col];
  return f2bf_rne(v);
}

__global__ __launch_bounds__(BLOCK) void prep_kernel(
    const float* __restrict__ weights,
    const float* __restrict__ W1, const float* __restrict__ b1,
    const float* __restrict__ W2, const float* __restrict__ b2,
    const float* __restrict__ W3, const float* __restrict__ b3,
    const float* __restrict__ W4, const float* __restrict__ b4,
    const int* __restrict__ data_idx, const int* __restrict__ edge_dir,
    char* __restrict__ ws) {
  const int tid = threadIdx.x;
  __bf16* wf1 = (__bf16*)(ws + WF1_OFF);
  __bf16* wf2 = (__bf16*)(ws + WF2_OFF);
  __bf16* wf3 = (__bf16*)(ws + WF3_OFF);
  float*  ftab = (float*)(ws + FTAB_OFF);
  int*    itab = (int*)(ws + ITAB_OFF);

  for (int e = tid; e < NT * 2 * 512; e += BLOCK) wf1[e] = frag_elem(W1, AROWS, 2, e);
  for (int e = tid; e < NT * 4 * 512; e += BLOCK) wf2[e] = frag_elem(W2, HREAL, 4, e);
  for (int e = tid; e < NT * 4 * 512; e += BLOCK) wf3[e] = frag_elem(W3, HREAL, 4, e);

  for (int k = tid; k < AROWS; k += BLOCK) {
    float g = weights[k] * (float)edge_dir[k];
    ftab[k] = g > 0.0f ? g : 0.0f;              // gate
    itab[k] = data_idx[k];                      // gather indices
  }
  for (int j = tid; j < HPAD; j += BLOCK) {
    ftab[64 + 0 * HPAD + j] = (j < HREAL) ? b1[j] : 0.0f;
    ftab[64 + 1 * HPAD + j] = (j < HREAL) ? b2[j] : 0.0f;
    ftab[64 + 2 * HPAD + j] = (j < HREAL) ? b3[j] : 0.0f;
    ftab[64 + 3 * HPAD + j] = (j < HREAL) ? W4[j] : 0.0f;
  }
  if (tid == 0) ftab[64 + 4 * HPAD] = b4[0];
}

// ---------------------------------------------------------------------------
// TDM: contiguous global -> LDS DMA of nbytes (multiple of 8), via the
// gfx1250 Tensor Data Mover.  Descriptor per CDNA5 ISA:
// group0 = {count=1, lds_addr, global_addr, type=2}; group1 = 1-row tile of
// 8-byte elements (data_size=3), tile_dim0 = tensor_dim0 = nbytes/8.
// This toolchain exposes the 6-arg builtin (extra int32x8 group, zeroed).
// ---------------------------------------------------------------------------
__device__ __forceinline__ void tdm_load_blob(const void* gsrc,
                                              unsigned lds_off,
                                              unsigned nbytes) {
  unsigned long long ga = (unsigned long long)gsrc;
  unsigned T = nbytes >> 3;                       // 8-byte units (fits 16 bits)
  v4u g0;
  g0[0] = 1u;                                     // count=1, user descriptor
  g0[1] = lds_off;                                // lds_addr (bytes)
  g0[2] = (unsigned)(ga & 0xFFFFFFFFu);           // global_addr[31:0]
  g0[3] = (unsigned)((ga >> 32) & 0x01FFFFFFu)    // global_addr[56:32]
          | 0x80000000u;                          // type = 2 ("image")
  v8i g1;
  g1[0] = (int)(3u << 16);                        // data_size=3 (8B), no mcast
  g1[1] = (int)((T & 0xFFFFu) << 16);             // tensor_dim0[15:0] @ bit48
  g1[2] = (int)(((T >> 16) & 0xFFFFu)             // tensor_dim0[31:16]
                | (1u << 16));                    // tensor_dim1 = 1
  g1[3] = (int)((T & 0xFFFFu) << 16);             // tile_dim0 @ bits 127:112
  g1[4] = 1;                                      // tile_dim1 = 1
  g1[5] = (int)T;                                 // tensor_dim0_stride lo
  g1[6] = 0;
  g1[7] = 0;
  v4i z4 = {0, 0, 0, 0};
  v8i z8 = {0, 0, 0, 0, 0, 0, 0, 0};
  __builtin_amdgcn_tensor_load_to_lds(g0, g1, z4, z4, z8, 0);
}

__device__ __forceinline__ v16bf load_bfrag(const __bf16* wbuf, int frag, int lane) {
  return *(const v16bf*)&wbuf[((frag * 32 + lane) << 4)];
}

__device__ __forceinline__ void hidden_layer(__bf16 (&x)[16][KPAD2],
                                             const __bf16* wbuf,
                                             const float* bias, int lane) {
  const int r    = lane & 15;
  const int half = lane >> 4;
  v16bf a[4];
#pragma unroll
  for (int f = 0; f < 4; ++f) {
    v8bf lo = *(const v8bf*)&x[r][f * 32 + half * 8];
    v8bf hi = *(const v8bf*)&x[r][f * 32 + 16 + half * 8];
    a[f] = __builtin_shufflevector(lo, hi, 0, 1, 2, 3, 4, 5, 6, 7,
                                           8, 9, 10, 11, 12, 13, 14, 15);
  }
#pragma unroll
  for (int nt = 0; nt < NT; ++nt) {
    v8f c = {};
#pragma unroll
    for (int ks = 0; ks < 4; ++ks) {
      c = __builtin_amdgcn_wmma_f32_16x16x32_bf16(
              false, a[ks], false, load_bfrag(wbuf, nt * 4 + ks, lane),
              (short)0, c, false, false);
    }
    const int col = nt * 16 + r;
    const float bv = bias[col];
#pragma unroll
    for (int j = 0; j < 8; ++j) {
      float v = c[j] + bv;
      x[j + 8 * half][col] = f2bf_t(v > 0.0f ? v : 0.0f);
    }
  }
}

__global__ __launch_bounds__(BLOCK) void nodal_mlp_kernel(
    const float* __restrict__ data,
    const char* __restrict__ ws,
    float* __restrict__ out, int Btot) {
  __shared__ __attribute__((aligned(32))) __bf16 wbuf[NT * 4 * 512];   // 28 KB, TDM-reloaded per layer
  __shared__ __attribute__((aligned(32))) __bf16 xs[WAVES][16][KPAD2]; // 32 KB staging
  __shared__ float gate_s[AROWS];
  __shared__ int   gidx_s[AROWS];
  __shared__ float bias_s[3 * HPAD];
  __shared__ float w4_s[HPAD];

  const float* ftab = (const float*)(ws + FTAB_OFF);
  const int*   itab = (const int*)(ws + ITAB_OFF);

  const int tid  = threadIdx.x;
  const int wave = tid >> 5;
  const int lane = tid & 31;
  const int r    = lane & 15;
  const int half = lane >> 4;
  const long row  = (long)blockIdx.x * (WAVES * 16) + wave * 16 + r;
  const bool rowOK = row < (long)Btot;
  const unsigned wbuf_lds = (unsigned)(unsigned long long)(&wbuf[0]);

  // ---- phase 0: kick off W1 DMA, stage tables, zero padding ----
  if (tid == 0) tdm_load_blob(ws + WF1_OFF, wbuf_lds, NT * 2 * 512 * 2);
  for (int k = tid; k < AROWS; k += BLOCK) {
    gate_s[k] = ftab[k];
    gidx_s[k] = itab[k];
  }
  for (int j = tid; j < 3 * HPAD; j += BLOCK) bias_s[j] = ftab[64 + j];
  for (int j = tid; j < HPAD; j += BLOCK)     w4_s[j]   = ftab[64 + 3 * HPAD + j];
  {
    __bf16* p = &xs[0][0][0];
    const __bf16 z = f2bf_t(0.0f);
    for (int e = tid; e < WAVES * 16 * KPAD2; e += BLOCK) p[e] = z;
  }
  __syncthreads();   // gate/gidx visible; TDM still in flight

  // ---- gather gated columns into A fragments (overlaps the W1 DMA) ----
  v16bf a1[2];
#pragma unroll
  for (int f = 0; f < 2; ++f) {
    A16 t;
#pragma unroll
    for (int i = 0; i < 16; ++i) {
      int kl = f * 32 + ((i < 8) ? (half * 8 + i) : (16 + half * 8 + (i - 8)));
      float v = 0.0f;
      if (rowOK) v = data[(size_t)row * 256 + gidx_s[kl]] * gate_s[kl];
      t.e[i] = f2bf_t(v);
    }
    a1[f] = __builtin_bit_cast(v16bf, t);
  }
  if (tid == 0) __builtin_amdgcn_s_wait_tensorcnt(0);
  __syncthreads();

  // ---- layer 1 ----
#pragma unroll
  for (int nt = 0; nt < NT; ++nt) {
    v8f c = {};
#pragma unroll
    for (int ks = 0; ks < 2; ++ks) {
      c = __builtin_amdgcn_wmma_f32_16x16x32_bf16(
              false, a1[ks], false, load_bfrag(wbuf, nt * 2 + ks, lane),
              (short)0, c, false, false);
    }
    const int col = nt * 16 + r;
    const float bv = bias_s[col];
#pragma unroll
    for (int j = 0; j < 8; ++j) {
      float v = c[j] + bv;
      xs[wave][j + 8 * half][col] = f2bf_t(v > 0.0f ? v : 0.0f);
    }
  }
  __syncthreads();   // all waves done reading W1 fragments

  // ---- layer 2: DMA W2 fragments, compute ----
  if (tid == 0) {
    tdm_load_blob(ws + WF2_OFF, wbuf_lds, NT * 4 * 512 * 2);
    __builtin_amdgcn_s_wait_tensorcnt(0);
  }
  __syncthreads();
  hidden_layer(xs[wave], wbuf, bias_s + HPAD, lane);
  __syncthreads();

  // ---- layer 3: DMA W3 fragments, compute ----
  if (tid == 0) {
    tdm_load_blob(ws + WF3_OFF, wbuf_lds, NT * 4 * 512 * 2);
    __builtin_amdgcn_s_wait_tensorcnt(0);
  }
  __syncthreads();
  hidden_layer(xs[wave], wbuf, bias_s + 2 * HPAD, lane);

  // ---- layer 4: 100 -> 1 dot product, two lanes per row ----
  float partial = 0.0f;
#pragma unroll
  for (int c8 = 0; c8 < 7; ++c8) {
    v8bf xv = *(const v8bf*)&xs[wave][r][half * 56 + c8 * 8];
#pragma unroll
    for (int i = 0; i < 8; ++i)
      partial += bf2f(xv[i]) * w4_s[half * 56 + c8 * 8 + i];
  }
  float other = __shfl_xor(partial, 16);
  partial += other;
  if (half == 0 && rowOK) out[row] = partial + ftab[64 + 4 * HPAD];
}

extern "C" void kernel_launch(void* const* d_in, const int* in_sizes, int n_in,
                              void* d_out, int out_size, void* d_ws, size_t ws_size,
                              hipStream_t stream) {
  (void)n_in; (void)ws_size; (void)out_size;
  const float* data     = (const float*)d_in[0];
  const float* weights  = (const float*)d_in[1];
  const float* W1       = (const float*)d_in[2];
  const float* b1       = (const float*)d_in[3];
  const float* W2       = (const float*)d_in[4];
  const float* b2       = (const float*)d_in[5];
  const float* W3       = (const float*)d_in[6];
  const float* b3       = (const float*)d_in[7];
  const float* W4       = (const float*)d_in[8];
  const float* b4       = (const float*)d_in[9];
  const int*   data_idx = (const int*)d_in[10];
  const int*   edge_dir = (const int*)d_in[11];
  float* out = (float*)d_out;
  char*  ws  = (char*)d_ws;

  const int Btot   = in_sizes[0] / 256;       // 500000
  const int blocks = (Btot + WAVES * 16 - 1) / (WAVES * 16);

  prep_kernel<<<1, BLOCK, 0, stream>>>(weights, W1, b1, W2, b2, W3, b3,
                                       W4, b4, data_idx, edge_dir, ws);
  nodal_mlp_kernel<<<blocks, BLOCK, 0, stream>>>(data, ws, out, Btot);
}
